// MambaLayer_1563368096407
// MI455X (gfx1250) — matrix operand order repaired and checked by reading, not compiled, not run on hardware
//
#include <hip/hip_runtime.h>
#include <hip/hip_bf16.h>
#include <math.h>

// Problem constants (from reference)
#define B_SZ      2
#define SEQ       4096            // 64*64
#define DIM_C     512
#define DI        1024            // D_INNER
#define NROWS     (B_SZ * SEQ)    // 8192
#define XPN       64              // DT_RANK + 2*D_STATE

typedef __attribute__((ext_vector_type(16))) __bf16 v16bf;
typedef __attribute__((ext_vector_type(8)))  __bf16 v8bf;
typedef __attribute__((ext_vector_type(8)))  float  v8f;

__device__ __forceinline__ float silu_f(float x) { return x / (1.0f + __expf(-x)); }

// ---------------------------------------------------------------------------
// f32 -> bf16 conversion (weights + activations staging)
// ---------------------------------------------------------------------------
__global__ void k_f32_to_bf16(const float* __restrict__ in, __bf16* __restrict__ out, int n) {
  int i = blockIdx.x * blockDim.x + threadIdx.x;
  if (i < n) out[i] = (__bf16)in[i];
}

// ---------------------------------------------------------------------------
// LayerNorm over C=512 per (b,l); one wave per position; writes bf16 rows
// x layout: (B, C, L)  ->  xn layout: (NROWS, 512) row-major bf16
// ---------------------------------------------------------------------------
__global__ void k_layernorm(const float* __restrict__ x, const float* __restrict__ w,
                            const float* __restrict__ b, __bf16* __restrict__ xn) {
  int pos  = blockIdx.x * (blockDim.x >> 5) + (threadIdx.x >> 5);
  int lane = threadIdx.x & 31;
  if (pos >= NROWS) return;
  int bb = pos >> 12;
  int l  = pos & (SEQ - 1);
  const float* xp = x + ((size_t)bb * DIM_C) * SEQ + l;

  float vals[16], s = 0.f, s2 = 0.f;
#pragma unroll
  for (int i = 0; i < 16; i++) {
    int c = lane + i * 32;
    float v = xp[(size_t)c * SEQ];
    vals[i] = v; s += v; s2 += v * v;
  }
#pragma unroll
  for (int off = 16; off > 0; off >>= 1) {
    s  += __shfl_xor(s,  off);
    s2 += __shfl_xor(s2, off);
  }
  float mu  = s  * (1.0f / DIM_C);
  float var = s2 * (1.0f / DIM_C) - mu * mu;
  float rs  = rsqrtf(var + 1e-5f);
#pragma unroll
  for (int i = 0; i < 16; i++) {
    int c = lane + i * 32;
    xn[(size_t)pos * DIM_C + c] = (__bf16)((vals[i] - mu) * rs * w[c] + b[c]);
  }
}

// ---------------------------------------------------------------------------
// Register-blocked WMMA macro-tile: out = A(MxK) * W(NxK)^T, bf16 in, f32 acc
// Each wave computes (MT*16) x (NT*16); per K-step of 32 it loads MT A-tiles
// + NT B-tiles and issues MT*NT wmma ops (A reused NT times, B reused MT).
//
// A-matrix lane layout (16-bit, 16x32): elems 0-7  -> K = k0 + half*8  + i
//                                       elems 8-15 -> K = k0 + 16 + half*8 + i
// B-matrix lane layout (16-bit, 32x16): same as A (elems 0-7 -> k0+half*8+i, 8-15 -> k0+16+half*8+i)
// ---------------------------------------------------------------------------
template <int MT, int NT>
__device__ __forceinline__ void wmma_block(const __bf16* __restrict__ A,
                                           const __bf16* __restrict__ Wt,
                                           int K, int mBase, int nBase,
                                           v8f acc[MT][NT]) {
  int lane = threadIdx.x & 31;
  int half = lane >> 4;
  int r16  = lane & 15;
#pragma unroll
  for (int mt = 0; mt < MT; mt++)
#pragma unroll
    for (int nt = 0; nt < NT; nt++)
      acc[mt][nt] = (v8f){0.f, 0.f, 0.f, 0.f, 0.f, 0.f, 0.f, 0.f};

  for (int k0 = 0; k0 < K; k0 += 32) {
    v16bf av[MT], bv[NT];
#pragma unroll
    for (int mt = 0; mt < MT; mt++) {
      const __bf16* ap = A + (size_t)(mBase + mt * 16 + r16) * K + k0 + half * 8;
      v8bf alo = *(const v8bf*)ap;
      v8bf ahi = *(const v8bf*)(ap + 16);
#pragma unroll
      for (int i = 0; i < 8; i++) { av[mt][i] = alo[i]; av[mt][i + 8] = ahi[i]; }
    }
#pragma unroll
    for (int nt = 0; nt < NT; nt++) {
      const __bf16* bp = Wt + (size_t)(nBase + nt * 16 + r16) * K + k0 + half * 8;
      v8bf blo = *(const v8bf*)bp;
      v8bf bhi = *(const v8bf*)(bp + 16);
#pragma unroll
      for (int i = 0; i < 8; i++) { bv[nt][i] = blo[i]; bv[nt][i + 8] = bhi[i]; }
    }
#pragma unroll
    for (int mt = 0; mt < MT; mt++)
#pragma unroll
      for (int nt = 0; nt < NT; nt++)
        acc[mt][nt] = __builtin_amdgcn_wmma_f32_16x16x32_bf16(
            false, av[mt], false, bv[nt], (short)0, acc[mt][nt], false, false);
  }
}

// Generic GEMM: out(MxN) = A * W^T
template <int MT, int NT>
__global__ void k_gemm(const __bf16* __restrict__ A, const __bf16* __restrict__ Wt,
                       float* __restrict__ out, int M, int N, int K) {
  int nBase = (blockIdx.x * (blockDim.x >> 5) + (threadIdx.x >> 5)) * (NT * 16);
  int mBase = blockIdx.y * (MT * 16);
  if (nBase >= N || mBase >= M) return;   // wave-uniform
  v8f acc[MT][NT];
  wmma_block<MT, NT>(A, Wt, K, mBase, nBase, acc);
  int lane = threadIdx.x & 31;
  int half = lane >> 4;
  int c16  = lane & 15;
#pragma unroll
  for (int mt = 0; mt < MT; mt++)
#pragma unroll
    for (int nt = 0; nt < NT; nt++)
#pragma unroll
      for (int r = 0; r < 8; r++)
        out[(size_t)(mBase + mt * 16 + half * 8 + r) * N + nBase + nt * 16 + c16] =
            acc[mt][nt][r];
}

// dt_proj GEMM with bias + softplus epilogue -> delta
template <int MT, int NT>
__global__ void k_gemm_softplus(const __bf16* __restrict__ A, const __bf16* __restrict__ Wt,
                                const float* __restrict__ bias, float* __restrict__ out,
                                int M, int N, int K) {
  int nBase = (blockIdx.x * (blockDim.x >> 5) + (threadIdx.x >> 5)) * (NT * 16);
  int mBase = blockIdx.y * (MT * 16);
  if (nBase >= N || mBase >= M) return;
  v8f acc[MT][NT];
  wmma_block<MT, NT>(A, Wt, K, mBase, nBase, acc);
  int lane = threadIdx.x & 31;
  int half = lane >> 4;
  int c16  = lane & 15;
#pragma unroll
  for (int nt = 0; nt < NT; nt++) {
    int col = nBase + nt * 16 + c16;
    float bs = bias[col];
#pragma unroll
    for (int mt = 0; mt < MT; mt++)
#pragma unroll
      for (int r = 0; r < 8; r++) {
        float v = acc[mt][nt][r] + bs;
        out[(size_t)(mBase + mt * 16 + half * 8 + r) * N + col] =
            (v > 20.f) ? v : log1pf(expf(v));
      }
  }
}

// out_proj GEMM with fused transpose store into (B, 512, 64, 64)
template <int MT, int NT>
__global__ void k_gemm_out(const __bf16* __restrict__ A, const __bf16* __restrict__ Wt,
                           float* __restrict__ dout, int M, int N, int K) {
  int nBase = (blockIdx.x * (blockDim.x >> 5) + (threadIdx.x >> 5)) * (NT * 16);
  int mBase = blockIdx.y * (MT * 16);
  if (nBase >= N || mBase >= M) return;
  v8f acc[MT][NT];
  wmma_block<MT, NT>(A, Wt, K, mBase, nBase, acc);
  int lane = threadIdx.x & 31;
  int half = lane >> 4;
  int c16  = lane & 15;
#pragma unroll
  for (int mt = 0; mt < MT; mt++)
#pragma unroll
    for (int nt = 0; nt < NT; nt++)
#pragma unroll
      for (int r = 0; r < 8; r++) {
        int row = mBase + mt * 16 + half * 8 + r;
        int bb  = row >> 12;
        int l   = row & (SEQ - 1);
        dout[((size_t)(bb * DIM_C + nBase + nt * 16 + c16)) * SEQ + l] = acc[mt][nt][r];
      }
}

// ---------------------------------------------------------------------------
// Depthwise causal conv(4) + SiLU. xz: (NROWS, 2048), xin = cols [0,1024)
// Writes xs (f32, scan input) and xsb (bf16, x_proj GEMM input)
// ---------------------------------------------------------------------------
__global__ void k_conv_silu(const float* __restrict__ xz, const float* __restrict__ cw,
                            const float* __restrict__ cb, float* __restrict__ xs,
                            __bf16* __restrict__ xsb) {
  size_t idx = (size_t)blockIdx.x * blockDim.x + threadIdx.x;
  if (idx >= (size_t)NROWS * DI) return;
  int d   = (int)(idx & (DI - 1));
  int row = (int)(idx >> 10);
  int l   = row & (SEQ - 1);
  int bb  = row >> 12;
  float acc = cb[d];
#pragma unroll
  for (int j = 0; j < 4; j++) {
    int ls = l - 3 + j;
    if (ls >= 0)
      acc += cw[d * 4 + j] * xz[((size_t)(bb * SEQ + ls)) * (2 * DI) + d];
  }
  float v = silu_f(acc);
  xs[idx]  = v;
  xsb[idx] = (__bf16)v;
}

// dt columns of x_dbl -> bf16 (A-matrix of dt_proj GEMM, K=32)
__global__ void k_dt_bf16(const float* __restrict__ xdbl, __bf16* __restrict__ dtb) {
  int i = blockIdx.x * blockDim.x + threadIdx.x;
  if (i >= NROWS * 32) return;
  int row = i >> 5, j = i & 31;
  dtb[i] = (__bf16)xdbl[(size_t)row * XPN + j];
}

// ---------------------------------------------------------------------------
// Selective scan: one thread per (b, d) row; 16 states in registers; A[d][:]
// loaded once. B/C read per step from x_dbl cols [32,64) (broadcast in wave).
// ---------------------------------------------------------------------------
__global__ void k_scan(const float* __restrict__ xs, const float* __restrict__ delta,
                       const float* __restrict__ xdbl, const float* __restrict__ A_log,
                       const float* __restrict__ Dk, float* __restrict__ y) {
  int d  = blockIdx.x * blockDim.x + threadIdx.x;
  int bb = blockIdx.y;
  if (d >= DI) return;
  float A[16];
#pragma unroll
  for (int n = 0; n < 16; n++) A[n] = -__expf(A_log[d * 16 + n]);
  float dsk = Dk[d];
  float h[16];
#pragma unroll
  for (int n = 0; n < 16; n++) h[n] = 0.f;
  for (int l = 0; l < SEQ; l++) {
    size_t row = (size_t)bb * SEQ + l;
    float u  = xs[row * DI + d];
    float dt = delta[row * DI + d];
    const float* bc = xdbl + row * XPN + 32;  // Bm[0..15], Cm[16..31]
    float du = dt * u;
    float yv = 0.f;
#pragma unroll
    for (int n = 0; n < 16; n++) {
      float dA = __expf(dt * A[n]);
      h[n] = dA * h[n] + du * bc[n];
      yv   = fmaf(h[n], bc[16 + n], yv);
    }
    y[row * DI + d] = yv + u * dsk;
  }
}

// Gate: yg = bf16( y * silu(z) ), z = xz cols [1024, 2048)
__global__ void k_gate(const float* __restrict__ y, const float* __restrict__ xz,
                       __bf16* __restrict__ yg) {
  size_t idx = (size_t)blockIdx.x * blockDim.x + threadIdx.x;
  if (idx >= (size_t)NROWS * DI) return;
  int d = (int)(idx & (DI - 1));
  size_t row = idx >> 10;
  float z = xz[row * (2 * DI) + DI + d];
  yg[idx] = (__bf16)(y[idx] * silu_f(z));
}

// ---------------------------------------------------------------------------
extern "C" void kernel_launch(void* const* d_in, const int* in_sizes, int n_in,
                              void* d_out, int out_size, void* d_ws, size_t ws_size,
                              hipStream_t stream) {
  const float* x        = (const float*)d_in[0];
  const float* ln_w     = (const float*)d_in[1];
  const float* ln_b     = (const float*)d_in[2];
  const float* in_projw = (const float*)d_in[3];
  const float* conv_w   = (const float*)d_in[4];
  const float* conv_b   = (const float*)d_in[5];
  const float* x_projw  = (const float*)d_in[6];
  const float* dt_projw = (const float*)d_in[7];
  const float* dt_projb = (const float*)d_in[8];
  const float* A_log    = (const float*)d_in[9];
  const float* D_skip   = (const float*)d_in[10];
  const float* out_projw= (const float*)d_in[11];
  float* out = (float*)d_out;

  char* wsb = (char*)d_ws;
  size_t off = 0;
  auto alloc = [&](size_t bytes) -> void* {
    off = (off + 255) & ~(size_t)255;
    void* p = wsb + off;
    off += bytes;
    return p;
  };

  __bf16* xn    = (__bf16*)alloc((size_t)NROWS * DIM_C * 2);
  __bf16* w_in  = (__bf16*)alloc((size_t)2 * DI * DIM_C * 2);
  float*  xz    = (float*) alloc((size_t)NROWS * 2 * DI * 4);
  float*  xs    = (float*) alloc((size_t)NROWS * DI * 4);
  __bf16* xsb   = (__bf16*)alloc((size_t)NROWS * DI * 2);
  __bf16* w_xp  = (__bf16*)alloc((size_t)XPN * DI * 2);
  float*  xdbl  = (float*) alloc((size_t)NROWS * XPN * 4);
  __bf16* dtb   = (__bf16*)alloc((size_t)NROWS * 32 * 2);
  __bf16* w_dt  = (__bf16*)alloc((size_t)DI * 32 * 2);
  float*  delta = (float*) alloc((size_t)NROWS * DI * 4);
  float*  yb    = (float*) alloc((size_t)NROWS * DI * 4);
  __bf16* ygb   = (__bf16*)alloc((size_t)NROWS * DI * 2);
  __bf16* w_out = (__bf16*)alloc((size_t)DIM_C * DI * 2);

  // Weight conversions (deterministic, redone each call)
  k_f32_to_bf16<<<(2 * DI * DIM_C + 255) / 256, 256, 0, stream>>>(in_projw,  w_in,  2 * DI * DIM_C);
  k_f32_to_bf16<<<(XPN * DI + 255) / 256,       256, 0, stream>>>(x_projw,   w_xp,  XPN * DI);
  k_f32_to_bf16<<<(DI * 32 + 255) / 256,        256, 0, stream>>>(dt_projw,  w_dt,  DI * 32);
  k_f32_to_bf16<<<(DIM_C * DI + 255) / 256,     256, 0, stream>>>(out_projw, w_out, DIM_C * DI);

  // 1) LayerNorm -> xn (bf16)
  k_layernorm<<<NROWS / 8, 256, 0, stream>>>(x, ln_w, ln_b, xn);

  // 2) in_proj: (8192x512) x (2048x512)^T -> xz (8192x2048)
  //    wave = 32x64 macro-tile; block = 4 waves along N = 32x256
  k_gemm<2, 4><<<dim3((2 * DI) / 256, NROWS / 32), 128, 0, stream>>>(
      xn, w_in, xz, NROWS, 2 * DI, DIM_C);

  // 3) depthwise causal conv + SiLU -> xs (f32) + xsb (bf16)
  k_conv_silu<<<(NROWS * DI) / 256, 256, 0, stream>>>(xz, conv_w, conv_b, xs, xsb);

  // 4) x_proj: (8192x1024) x (64x1024)^T -> x_dbl (8192x64); N small -> NT=1
  k_gemm<2, 1><<<dim3(1, NROWS / 32), 128, 0, stream>>>(
      xsb, w_xp, xdbl, NROWS, XPN, DI);

  // 5) dt -> bf16; dt_proj + softplus -> delta (8192x1024)
  k_dt_bf16<<<(NROWS * 32) / 256, 256, 0, stream>>>(xdbl, dtb);
  k_gemm_softplus<2, 4><<<dim3(DI / 256, NROWS / 32), 128, 0, stream>>>(
      dtb, w_dt, dt_projb, delta, NROWS, DI, 32);

  // 6) selective scan -> y (8192x1024)
  k_scan<<<dim3(DI / 256, B_SZ), 256, 0, stream>>>(xs, delta, xdbl, A_log, D_skip, yb);

  // 7) gate with silu(z) -> bf16
  k_gate<<<(NROWS * DI) / 256, 256, 0, stream>>>(yb, xz, ygb);

  // 8) out_proj with fused transpose store -> (2, 512, 64, 64)
  k_gemm_out<2, 4><<<dim3(DIM_C / 256, NROWS / 32), 128, 0, stream>>>(
      ygb, w_out, out, NROWS, DIM_C, DI);
}
